// RandomFeaturesLayer_36756330120125
// MI455X (gfx1250) — matrix operand order repaired and checked by reading, compiled, not run on hardware
//
#include <hip/hip_runtime.h>
#include <hip/hip_bf16.h>

// RandomFeaturesLayer: out = (x @ rp^T) @ w^T + 0.1*bias
// B=4096, IN=2048, OUT=2048, fp32 in/out. bf16 WMMA, f32 accumulate.
// Pass 0: rp, w -> bf16 in workspace (one-shot)
// Pass 1: h_bf16 = x @ rp16^T           (x pipelined through registers + cvt)
// Pass 2: out = h_bf16 @ w16^T + bias
// bf16 tiles staged via TDM tensor_load_to_lds (descriptor DMA, HW row padding),
// falling back to per-lane async-to-LDS, then to sync copies. Double-buffered LDS.

typedef __attribute__((ext_vector_type(16))) __bf16        v16bf;
typedef __attribute__((ext_vector_type(8)))  float         v8f;
typedef __attribute__((ext_vector_type(4)))  unsigned int  uint4v;
typedef __attribute__((ext_vector_type(4)))  int           int4v;
typedef __attribute__((ext_vector_type(8)))  int           int8v;

#define BM 128
#define BN 128
#define BK 32
#define LDST 20               // LDS row stride in uints (80 B = 64 B row + 16 B pad)
#define TILEU (BM * LDST)     // uints per tile buffer (10 KB)

#if __has_builtin(__builtin_amdgcn_tensor_load_to_lds) && \
    __has_builtin(__builtin_amdgcn_s_wait_tensorcnt)
#define HAVE_TDM 1
#else
#define HAVE_TDM 0
#endif

#if __has_builtin(__builtin_amdgcn_global_load_async_to_lds_b128)
#define HAVE_ASYNC_LDS 1
typedef __attribute__((address_space(1))) int4v* gvec_ptr;
typedef __attribute__((address_space(3))) int4v* lvec_ptr;
#else
#define HAVE_ASYNC_LDS 0
#endif

union FragCast { uint4v u[2]; v16bf v; };

static __device__ __forceinline__ unsigned int f2bf1(float f) {
  unsigned int u = __float_as_uint(f);
  return (u + 0x7FFFu + ((u >> 16) & 1u)) >> 16;   // RNE fp32->bf16
}
static __device__ __forceinline__ unsigned int pkbf(float lo, float hi) {
  return f2bf1(lo) | (f2bf1(hi) << 16);
}

static __device__ __forceinline__ unsigned int lds_offset_bytes(const unsigned int* p) {
  return (unsigned int)(uintptr_t)(__attribute__((address_space(3))) const void*)p;
}

static __device__ __forceinline__ void copy16_to_lds(const unsigned short* g,
                                                     unsigned int* l) {
#if HAVE_ASYNC_LDS
  __builtin_amdgcn_global_load_async_to_lds_b128(
      (gvec_ptr)(uintptr_t)g, (lvec_ptr)l, 0, 0);
#else
  *(uint4v*)l = *(const uint4v*)g;
#endif
}

#if HAVE_TDM
// One-shot descriptor DMA: 128-row x 32-element bf16 tile, row stride K elems,
// HW-padded in LDS to 80 B/row (pad_interval=16 DW, pad_amount=4 DW).
static __device__ __forceinline__ void tdm_load_tile(const unsigned short* g,
                                                     unsigned int lds_off, int K) {
  unsigned long long ga = (unsigned long long)(uintptr_t)g;
  uint4v g0;
  g0[0] = 1u;                                       // count=1, user D#
  g0[1] = lds_off;                                  // lds_addr (bytes)
  g0[2] = (unsigned int)ga;                         // global_addr[31:0]
  g0[3] = (unsigned int)(ga >> 32) | (2u << 30);    // global_addr[56:32] | type=2
  int8v g1;
  g1[0] = (int)((1u << 16)        // data_size = 2 bytes
              | (1u << 20)        // pad_enable
              | (3u << 22)        // pad_interval: 16 DWORDs (one 64 B row)
              | (3u << 25));      // pad_amount: 4 DWORDs (16 B)
  g1[1] = (int)(((unsigned)K & 0xFFFFu) << 16);         // tensor_dim0[15:0]
  g1[2] = (int)((((unsigned)K >> 16) & 0xFFFFu)         // tensor_dim0[31:16]
              | ((unsigned)BM << 16));                  // tensor_dim1[15:0] = 128
  g1[3] = (int)((unsigned)BK << 16);                    // dim1 hi=0 | tile_dim0 = 32
  g1[4] = (int)(unsigned)BM;                            // tile_dim1 = 128 | tile_dim2 = 0
  g1[5] = (int)K;                                       // tensor_dim0_stride[31:0]
  g1[6] = 0;                                            // stride hi | dim1_stride lo
  g1[7] = 0;
  int4v z4 = {0, 0, 0, 0};                              // groups 2/3 unused (2D tile)
  int8v z8 = {0, 0, 0, 0, 0, 0, 0, 0};
  __builtin_amdgcn_tensor_load_to_lds(g0, g1, z4, z4, z8, 0);
}
#endif

// Stage one bf16 tile (128 x 32) into LDS buffer lbuf.
static __device__ __forceinline__ void stage_bf16_tile(const unsigned short* g, int K,
                                                       unsigned int* lbuf, int wave,
                                                       const int rowC[2], const int cC[2]) {
#if HAVE_TDM
  if (wave == 0) tdm_load_tile(g, lds_offset_bytes(lbuf), K);
#else
  #pragma unroll
  for (int i = 0; i < 2; ++i)
    copy16_to_lds(g + (size_t)rowC[i] * K + cC[i] * 8,
                  &lbuf[rowC[i] * LDST + cC[i] * 4]);
#endif
}

static __device__ __forceinline__ void wait_staging() {
#if HAVE_TDM
  __builtin_amdgcn_s_wait_tensorcnt(0);
#elif HAVE_ASYNC_LDS
#if __has_builtin(__builtin_amdgcn_s_wait_asynccnt)
  __builtin_amdgcn_s_wait_asynccnt(0);
#else
  asm volatile("s_wait_asynccnt 0x0" ::: "memory");
#endif
#endif
}

// ---------------- one-shot fp32 -> bf16 conversion ----------------
__global__ __launch_bounds__(256) void cvt_to_bf16(const float* __restrict__ src,
                                                   unsigned short* __restrict__ dst,
                                                   int n4) {
  int i = blockIdx.x * 256 + threadIdx.x;
  if (i < n4) {
    float4 v = ((const float4*)src)[i];
    uint2 p; p.x = pkbf(v.x, v.y); p.y = pkbf(v.z, v.w);
    ((uint2*)dst)[i] = p;
  }
}

// ---------------- TN GEMM: C[M,N] = A[M,K] * B16[N,K]^T ----------------
// SECOND=false: A fp32 (x), C bf16 (h). SECOND=true: A bf16 (h), C fp32 + 0.1*bias.
template<bool SECOND>
__global__ __launch_bounds__(256)
void rf_gemm(const void* __restrict__ Av, const unsigned short* __restrict__ B16,
             const float* __restrict__ bias, void* __restrict__ Cv,
             int M, int N, int K)
{
  __shared__ unsigned int ldsA[2 * TILEU];
  __shared__ unsigned int ldsB[2 * TILEU];

  const int tid  = threadIdx.x;
  const int lane = tid & 31;
  const int wave = tid >> 5;
  const int wr   = wave >> 1;    // 4 waves along M (32 rows each)
  const int wc   = wave & 1;     // 2 waves along N (64 cols each)
  const int m0   = blockIdx.y * BM;
  const int n0   = blockIdx.x * BN;
  const int lm   = lane & 15;
  const int lh   = lane >> 4;

  const float*          Af = (const float*)Av;
  const unsigned short* Ah = (const unsigned short*)Av;

  // staging geometry (loop-invariant)
  int rowC[2], cC[2];            // bf16 16B chunks (async fallback): 2/thread
  #pragma unroll
  for (int i = 0; i < 2; ++i) { int c = tid + i * 256; rowC[i] = c >> 2; cC[i] = c & 3; }
  int rowF[4], cF[4];            // fp32 float4 chunks: 4/thread
  #pragma unroll
  for (int i = 0; i < 4; ++i) { int c = tid + i * 256; rowF[i] = c >> 3; cF[i] = c & 7; }

  v8f acc[2][4] = {};

  // ---- prologue: stage tile 0 into buffer 0 ----
  if constexpr (!SECOND) {
    #pragma unroll
    for (int i = 0; i < 4; ++i) {
      float4 v = *(const float4*)(Af + (size_t)(m0 + rowF[i]) * K + cF[i] * 4);
      int b = rowF[i] * LDST + cF[i] * 2;
      ldsA[b + 0] = pkbf(v.x, v.y);
      ldsA[b + 1] = pkbf(v.z, v.w);
    }
  } else {
    stage_bf16_tile(Ah + (size_t)m0 * K, K, &ldsA[0], wave, rowC, cC);
  }
  stage_bf16_tile(B16 + (size_t)n0 * K, K, &ldsB[0], wave, rowC, cC);
  wait_staging();
  __syncthreads();

  int p = 0;
  for (int k0 = 0; k0 < K; k0 += BK) {
    const int  kn       = k0 + BK;
    const bool has_next = kn < K;        // uniform
    const int  q        = p ^ 1;

    // ---- issue tile k+1 staging into buffer q (freed by last barrier) ----
    float4 aReg[4];
    if (has_next) {
      if constexpr (!SECOND) {
        #pragma unroll
        for (int i = 0; i < 4; ++i)
          aReg[i] = *(const float4*)(Af + (size_t)(m0 + rowF[i]) * K + kn + cF[i] * 4);
      } else {
        stage_bf16_tile(Ah + (size_t)m0 * K + kn, K, &ldsA[q * TILEU], wave, rowC, cC);
      }
      stage_bf16_tile(B16 + (size_t)n0 * K + kn, K, &ldsB[q * TILEU], wave, rowC, cC);
    }

    // ---- fragments from buffer p (CDNA5 WMMA VGPR layouts) ----
    FragCast a[2], b[4];
    const unsigned int* pA = &ldsA[p * TILEU];
    const unsigned int* pB = &ldsB[p * TILEU];
    #pragma unroll
    for (int mi = 0; mi < 2; ++mi) {
      const unsigned int* base = pA + (wr * 32 + mi * 16 + lm) * LDST;
      a[mi].u[0] = *(const uint4v*)(base + lh * 4);       // K = 8*lh .. +7
      a[mi].u[1] = *(const uint4v*)(base + 8 + lh * 4);   // K = 16+8*lh .. +7
    }
    #pragma unroll
    for (int ni = 0; ni < 4; ++ni) {
      const unsigned int* base = pB + (wc * 64 + ni * 16 + lm) * LDST;
      b[ni].u[0] = *(const uint4v*)(base + lh * 4);       // K = 8*lh .. +7
      b[ni].u[1] = *(const uint4v*)(base + 8 + lh * 4);   // K = 16+8*lh .. +7
    }

    // ---- 8 WMMAs: 32x64 sub-tile, K advance 32 ----
    #pragma unroll
    for (int mi = 0; mi < 2; ++mi)
      #pragma unroll
      for (int ni = 0; ni < 4; ++ni)
        acc[mi][ni] = __builtin_amdgcn_wmma_f32_16x16x32_bf16(
            false, a[mi].v, false, b[ni].v, (short)0, acc[mi][ni], false, false);

    // ---- finish tile k+1 staging, single barrier per iteration ----
    if (has_next) {
      if constexpr (!SECOND) {
        #pragma unroll
        for (int i = 0; i < 4; ++i) {            // load latency hidden by WMMAs above
          int bofs = q * TILEU + rowF[i] * LDST + cF[i] * 2;
          ldsA[bofs + 0] = pkbf(aReg[i].x, aReg[i].y);
          ldsA[bofs + 1] = pkbf(aReg[i].z, aReg[i].w);
        }
      }
      wait_staging();
      __syncthreads();
      p = q;
    }
  }

  // ---- epilogue: C/D layout VGPR r -> row 8*lh+r, col lm ----
  #pragma unroll
  for (int ni = 0; ni < 4; ++ni) {
    int gn = n0 + wc * 64 + ni * 16 + lm;
    float bterm = 0.0f;
    if (SECOND) bterm = 0.1f * bias[gn];
    #pragma unroll
    for (int mi = 0; mi < 2; ++mi) {
      #pragma unroll
      for (int r = 0; r < 8; ++r) {
        int gm = m0 + wr * 32 + mi * 16 + lh * 8 + r;
        if (SECOND) {
          ((float*)Cv)[(size_t)gm * N + gn] = acc[mi][ni][r] + bterm;
        } else {
          ((unsigned short*)Cv)[(size_t)gm * N + gn] =
              (unsigned short)f2bf1(acc[mi][ni][r]);
        }
      }
    }
  }
}

extern "C" void kernel_launch(void* const* d_in, const int* in_sizes, int n_in,
                              void* d_out, int out_size, void* d_ws, size_t ws_size,
                              hipStream_t stream) {
  constexpr int Bsz = 4096, IN = 2048, OUT = 2048;
  const float* x    = (const float*)d_in[0];   // [B, IN]
  const float* rp   = (const float*)d_in[1];   // [OUT, IN]
  const float* w    = (const float*)d_in[2];   // [OUT, OUT]
  const float* bias = (const float*)d_in[3];   // [1, OUT]
  float* out = (float*)d_out;                  // [B, OUT]

  // workspace (bf16 elements): rp16 [OUT*IN] | w16 [OUT*OUT] | h16 [B*OUT]
  unsigned short* rp16 = (unsigned short*)d_ws;
  unsigned short* w16  = rp16 + (size_t)OUT * IN;
  unsigned short* h16  = w16  + (size_t)OUT * OUT;

  const int n4 = OUT * IN / 4;
  cvt_to_bf16<<<(n4 + 255) / 256, 256, 0, stream>>>(rp, rp16, n4);
  cvt_to_bf16<<<(n4 + 255) / 256, 256, 0, stream>>>(w,  w16,  n4);

  dim3 grid(OUT / BN, Bsz / BM);               // (16, 32)
  rf_gemm<false><<<grid, 256, 0, stream>>>(x,   rp16, nullptr, h16, Bsz, OUT, IN);
  rf_gemm<true ><<<grid, 256, 0, stream>>>(h16, w16,  bias,    out, Bsz, OUT, OUT);
}
